// Recurrence_13108240187746
// MI455X (gfx1250) — compile-verified, matrix operand order checked by reading, not run on hardware
//
#include <hip/hip_runtime.h>
#include <hip/hip_bf16.h>

typedef unsigned short u16;
typedef __attribute__((ext_vector_type(16))) __bf16 v16bf;
typedef __attribute__((ext_vector_type(8)))  float  v8f;

#define T_STEPS 128
#define NB      512
#define HID     512
#define COND    64
#define N_LINES 10
#define NT      20
#define N_A     16
#define STATE   530      // 1 + 1 + HID + N_A
#define IN_ROW  130
#define G3      1536     // 3*HID
#define WIH_LD  1088     // IN_SIZE

__device__ __forceinline__ u16 f2bf(float f) {
  unsigned u = __float_as_uint(f);
  unsigned r = u + 0x7FFFu + ((u >> 16) & 1u);
  return (u16)(r >> 16);
}
__device__ __forceinline__ float bf2f(u16 h) {
  return __uint_as_float(((unsigned)h) << 16);
}
__device__ __forceinline__ float sigm(float x) { return 1.0f / (1.0f + __expf(-x)); }

union FragU { uint4 q[2]; v16bf v; u16 s[16]; };

// Load a 16x32 bf16 A/B fragment from a row-major tile. ld multiple of 8, rows 16B aligned.
// Lane L<16: row M=L, K = {0..7, 16..23}; lane L>=16: row M=L-16, K = {8..15, 24..31}.
__device__ __forceinline__ v16bf load_frag(const u16* base, int ld, int lane) {
  int r = lane & 15, hh = (lane >> 4) & 1;
  const u16* p = base + r * ld + hh * 8;
  FragU f;
  f.q[0] = *(const uint4*)(p);
  f.q[1] = *(const uint4*)(p + 16);
  return f.v;
}

// Same fragment layout, but source is fp32 with arbitrary ld (used for the cond slice of inputs).
__device__ __forceinline__ v16bf load_frag_f32(const float* base, int ld, int lane) {
  int r = lane & 15, hh = (lane >> 4) & 1;
  const float* p = base + r * ld + hh * 8;
  FragU f;
#pragma unroll
  for (int i = 0; i < 8; ++i) {
    f.s[i]     = f2bf(p[i]);
    f.s[i + 8] = f2bf(p[i + 16]);
  }
  return f.v;
}

__device__ __forceinline__ v8f wmma_bf16(v16bf a, v16bf b, v8f c) {
  return __builtin_amdgcn_wmma_f32_16x16x32_bf16(false, a, false, b, (short)0, c, false, false);
}

// ---------------------------------------------------------------------------
// Fused GRU step: gh = h @ whh.T (WMMA, K=512); optional gi-cond GEMM (K=64);
// gate fusion fully register-local per wave (each wave owns column triples).
// grid = 32 blocks (16 rows each), block = 256 threads = 8 waves.
// ---------------------------------------------------------------------------
__global__ __launch_bounds__(256) void gru_step_kernel(
    const u16* __restrict__ h_in,        // NB x HID bf16
    u16* __restrict__ h_out,             // NB x HID bf16
    const u16* __restrict__ whh,         // G3 x HID bf16
    const float* __restrict__ bhh,       // G3
    const float* __restrict__ gi_table,  // enc: NT x G3 (task_gi) ; main: NB x G3 (giH)
    const int* __restrict__ tok,         // enc: token ptr (pre-offset by step), stride N_LINES; null->main
    const float* __restrict__ xc,        // main: inputs + t*NB*IN_ROW (cond at col 0)
    const u16* __restrict__ wih,         // main: gru_wih bf16, ld WIH_LD (cond part = cols 0..63)
    float* __restrict__ hout_f32)        // main: d_out + t*NB*STATE + 2 (stride STATE); may be null
{
  const int m0 = blockIdx.x * 16;
  const int wv = threadIdx.x >> 5, lane = threadIdx.x & 31;
  const int nn_ = lane & 15, mo = (lane >> 4) * 8;
  const bool is_main = (tok == nullptr);

  for (int jt = wv; jt < 32; jt += 8) {
    v8f gh0 = {}, gh1 = {}, gh2 = {};
    for (int k0 = 0; k0 < HID; k0 += 32) {
      v16bf a  = load_frag(h_in + (size_t)m0 * HID + k0, HID, lane);
      v16bf b0 = load_frag(whh + (size_t)(0 * HID + jt * 16) * HID + k0, HID, lane);
      v16bf b1 = load_frag(whh + (size_t)(1 * HID + jt * 16) * HID + k0, HID, lane);
      v16bf b2 = load_frag(whh + (size_t)(2 * HID + jt * 16) * HID + k0, HID, lane);
      gh0 = wmma_bf16(a, b0, gh0);
      gh1 = wmma_bf16(a, b1, gh1);
      gh2 = wmma_bf16(a, b2, gh2);
    }
    v8f gi0 = {}, gi1 = {}, gi2 = {};
    if (is_main) {
#pragma unroll
      for (int k0 = 0; k0 < COND; k0 += 32) {
        v16bf a  = load_frag_f32(xc + (size_t)m0 * IN_ROW + k0, IN_ROW, lane);
        v16bf b0 = load_frag(wih + (size_t)(0 * HID + jt * 16) * WIH_LD + k0, WIH_LD, lane);
        v16bf b1 = load_frag(wih + (size_t)(1 * HID + jt * 16) * WIH_LD + k0, WIH_LD, lane);
        v16bf b2 = load_frag(wih + (size_t)(2 * HID + jt * 16) * WIH_LD + k0, WIH_LD, lane);
        gi0 = wmma_bf16(a, b0, gi0);
        gi1 = wmma_bf16(a, b1, gi1);
        gi2 = wmma_bf16(a, b2, gi2);
      }
    }
    const int j = jt * 16 + nn_;
    const float bh0 = bhh[j], bh1 = bhh[HID + j], bh2 = bhh[2 * HID + j];
#pragma unroll
    for (int r = 0; r < 8; ++r) {
      const int m = m0 + mo + r;
      float ir, iz, inx;
      if (is_main) {
        const float* g = gi_table + (size_t)m * G3;
        ir = g[j] + gi0[r]; iz = g[HID + j] + gi1[r]; inx = g[2 * HID + j] + gi2[r];
      } else {
        const int tk = tok[m * N_LINES];
        const float* g = gi_table + (size_t)tk * G3;
        ir = g[j]; iz = g[HID + j]; inx = g[2 * HID + j];
      }
      float rr = sigm(ir + gh0[r] + bh0);
      float uu = sigm(iz + gh1[r] + bh1);
      float nv = tanhf(inx + rr * (gh2[r] + bh2));
      float ho = bf2f(h_in[(size_t)m * HID + j]);
      float hv = (1.0f - uu) * nv + uu * ho;
      h_out[(size_t)m * HID + j] = f2bf(hv);
      if (hout_f32) hout_f32[(size_t)m * STATE + j] = hv;
    }
  }
}

// ---------------------------------------------------------------------------
// giH = H(NB x 1024) @ gru_wih[:,64:].T + gru_bih  -> NB x G3 fp32 (once).
// ---------------------------------------------------------------------------
__global__ __launch_bounds__(256) void giH_kernel(
    const u16* __restrict__ Hc, const u16* __restrict__ wih,
    const float* __restrict__ bih, float* __restrict__ giH)
{
  const int m0 = blockIdx.x * 16;
  const int wv = threadIdx.x >> 5, lane = threadIdx.x & 31;
  const int nn_ = lane & 15, mo = (lane >> 4) * 8;
  for (int jt = wv; jt < 96; jt += 8) {
    v8f acc = {};
    for (int k0 = 0; k0 < 1024; k0 += 32) {
      v16bf a = load_frag(Hc + (size_t)m0 * 1024 + k0, 1024, lane);
      v16bf b = load_frag(wih + (size_t)(jt * 16) * WIH_LD + COND + k0, WIH_LD, lane);
      acc = wmma_bf16(a, b, acc);
    }
    const int j = jt * 16 + nn_;
    const float bj = bih[j];
#pragma unroll
    for (int r = 0; r < 8; ++r)
      giH[(size_t)(m0 + mo + r) * G3 + j] = acc[r] + bj;
  }
}

// ---------------------------------------------------------------------------
// MLP (2 layers, relu) + actor/critic head + softmax + gumbel-argmax sample.
// One block per 16 rows; z1/z2 staged in LDS as bf16.
// ---------------------------------------------------------------------------
__global__ __launch_bounds__(256) void mlp_head_kernel(
    const u16* __restrict__ h,
    const u16* __restrict__ w1, const float* __restrict__ b1,
    const u16* __restrict__ w2, const float* __restrict__ b2,
    const u16* __restrict__ aw, const float* __restrict__ ab,
    const float* __restrict__ cw, const float* __restrict__ cb,
    const float* __restrict__ act_in,   // inputs + t*NB*IN_ROW + 74 (stride IN_ROW)
    float* __restrict__ out_t,          // d_out + t*NB*STATE
    int t)
{
  __shared__ alignas(16) u16 s_z1[16 * HID];
  __shared__ alignas(16) u16 s_z2[16 * HID];
  __shared__ float s_logit[16][N_A];
  __shared__ float s_v[16];

  const int m0 = blockIdx.x * 16;
  const int wv = threadIdx.x >> 5, lane = threadIdx.x & 31;
  const int nn_ = lane & 15, mo = (lane >> 4) * 8;

  // layer 1: z1 = relu(h @ w1.T + b1)
  for (int jt = wv; jt < 32; jt += 8) {
    v8f acc = {};
    for (int k0 = 0; k0 < HID; k0 += 32) {
      v16bf a = load_frag(h + (size_t)m0 * HID + k0, HID, lane);
      v16bf b = load_frag(w1 + (size_t)(jt * 16) * HID + k0, HID, lane);
      acc = wmma_bf16(a, b, acc);
    }
    const int j = jt * 16 + nn_;
    const float bj = b1[j];
#pragma unroll
    for (int r = 0; r < 8; ++r) {
      float z = acc[r] + bj;
      s_z1[(mo + r) * HID + j] = f2bf(z > 0.0f ? z : 0.0f);
    }
  }
  __syncthreads();

  // layer 2: z2 = relu(z1 @ w2.T + b2)
  for (int jt = wv; jt < 32; jt += 8) {
    v8f acc = {};
    for (int k0 = 0; k0 < HID; k0 += 32) {
      v16bf a = load_frag(s_z1 + k0, HID, lane);
      v16bf b = load_frag(w2 + (size_t)(jt * 16) * HID + k0, HID, lane);
      acc = wmma_bf16(a, b, acc);
    }
    const int j = jt * 16 + nn_;
    const float bj = b2[j];
#pragma unroll
    for (int r = 0; r < 8; ++r) {
      float z = acc[r] + bj;
      s_z2[(mo + r) * HID + j] = f2bf(z > 0.0f ? z : 0.0f);
    }
  }
  __syncthreads();

  // actor logits (wave 0, one 16x16 tile) ; critic dot (wave 1, one row per lane)
  if (wv == 0) {
    v8f acc = {};
    for (int k0 = 0; k0 < HID; k0 += 32) {
      v16bf a = load_frag(s_z2 + k0, HID, lane);
      v16bf b = load_frag(aw + k0, HID, lane);
      acc = wmma_bf16(a, b, acc);
    }
#pragma unroll
    for (int r = 0; r < 8; ++r) s_logit[mo + r][nn_] = acc[r] + ab[nn_];
  } else if (wv == 1 && lane < 16) {
    float v = cb[0];
    for (int k = 0; k < HID; ++k) v += bf2f(s_z2[lane * HID + k]) * cw[k];
    s_v[lane] = v;
  }
  __syncthreads();

  // softmax + deterministic gumbel-argmax sample + output row
  if (threadIdx.x < 16) {
    const int row = threadIdx.x;
    const int ng = m0 + row;
    float lg[N_A], mx = -1e30f;
#pragma unroll
    for (int k = 0; k < N_A; ++k) { lg[k] = s_logit[row][k]; mx = fmaxf(mx, lg[k]); }
    float se = 0.0f;
#pragma unroll
    for (int k = 0; k < N_A; ++k) { lg[k] = __expf(lg[k] - mx); se += lg[k]; }
    const float inv = 1.0f / se;

    int samp = 0; float best = -1e30f;
#pragma unroll
    for (int k = 0; k < N_A; ++k) {
      unsigned x = (unsigned)(t * 912367) ^ (unsigned)(ng * 2654435761u) ^ (unsigned)(k * 40503) ^ 0x9E3779B9u;
      x = x * 747796405u + 2891336453u;
      unsigned w = ((x >> ((x >> 28) + 4u)) ^ x) * 277803737u;
      w = (w >> 22) ^ w;
      float u = ((float)(w >> 8) + 0.5f) * (1.0f / 16777216.0f);
      float g = -__logf(-__logf(u));
      float sc = s_logit[row][k] + g;
      if (sc > best) { best = sc; samp = k; }
    }
    float a_prev = act_in[(size_t)ng * IN_ROW];
    float a = (a_prev < 0.0f) ? (float)samp : a_prev;
    float* o = out_t + (size_t)ng * STATE;
    o[0] = a;
    o[1] = s_v[row];
#pragma unroll
    for (int k = 0; k < N_A; ++k) o[2 + HID + k] = lg[k] * inv;
  }
}

// --------------------------- setup / small kernels -------------------------
__global__ void cvt_bf16_kernel(const float* __restrict__ in, u16* __restrict__ o, int n) {
  int i = blockIdx.x * blockDim.x + threadIdx.x;
  if (i < n) o[i] = f2bf(in[i]);
}
__global__ void fill0_kernel(u16* p, int n) {
  int i = blockIdx.x * blockDim.x + threadIdx.x;
  if (i < n) p[i] = 0;
}
__global__ void lines_kernel(const float* __restrict__ inputs, int* __restrict__ lines) {
  int i = blockIdx.x * blockDim.x + threadIdx.x;
  if (i < NB * N_LINES) {
    int nrow = i / N_LINES, l = i % N_LINES;
    lines[i] = (int)inputs[(size_t)nrow * IN_ROW + COND + l];
  }
}
__global__ void h0_kernel(const float* __restrict__ rnn, u16* __restrict__ h) {
  int i = blockIdx.x * blockDim.x + threadIdx.x;
  if (i < NB * HID) {
    int nrow = i >> 9, j = i & (HID - 1);
    h[i] = f2bf(rnn[(size_t)nrow * STATE + 2 + j]);
  }
}
__global__ void copy_half_kernel(const u16* __restrict__ src, u16* __restrict__ Hc, int off) {
  int i = blockIdx.x * blockDim.x + threadIdx.x;
  if (i < NB * HID) {
    int nrow = i >> 9, j = i & (HID - 1);
    Hc[(size_t)nrow * 1024 + off + j] = src[i];
  }
}
__global__ void task_gi_kernel(const float* __restrict__ emb, const float* __restrict__ wih,
                               const float* __restrict__ bih, float* __restrict__ outp) {
  int i = blockIdx.x * blockDim.x + threadIdx.x;
  if (i >= NT * G3) return;
  int task = i / G3, j = i % G3;
  const float* e = emb + (size_t)task * HID;
  const float* w = wih + (size_t)j * HID;
  float s = bih[j];
  for (int k = 0; k < HID; ++k) s += e[k] * w[k];
  outp[i] = s;
}
__global__ void tail_kernel(float* __restrict__ out) {
  int i = blockIdx.x * blockDim.x + threadIdx.x;
  if (i < NB * STATE)
    out[(size_t)T_STEPS * NB * STATE + i] = out[(size_t)(T_STEPS - 1) * NB * STATE + i];
}

// ---------------------------------------------------------------------------
extern "C" void kernel_launch(void* const* d_in, const int* in_sizes, int n_in,
                              void* d_out, int out_size, void* d_ws, size_t ws_size,
                              hipStream_t stream) {
  (void)in_sizes; (void)n_in; (void)out_size; (void)ws_size;
  const float* inputs  = (const float*)d_in[0];
  const float* rnn_hxs = (const float*)d_in[1];
  const float* embed   = (const float*)d_in[2];
  const float* ewih_f  = (const float*)d_in[3];
  const float* ewhh_f  = (const float*)d_in[4];
  const float* ebih_f  = (const float*)d_in[5];
  const float* ebhh_f  = (const float*)d_in[6];
  const float* ewih_b  = (const float*)d_in[7];
  const float* ewhh_b  = (const float*)d_in[8];
  const float* ebih_b  = (const float*)d_in[9];
  const float* ebhh_b  = (const float*)d_in[10];
  const float* gwih    = (const float*)d_in[11];
  const float* gwhh    = (const float*)d_in[12];
  const float* gbih    = (const float*)d_in[13];
  const float* gbhh    = (const float*)d_in[14];
  const float* mlpw    = (const float*)d_in[15];
  const float* mlpb    = (const float*)d_in[16];
  const float* actw    = (const float*)d_in[17];
  const float* actb    = (const float*)d_in[18];
  const float* crw     = (const float*)d_in[19];
  const float* crb     = (const float*)d_in[20];
  float* out = (float*)d_out;

  char* ws = (char*)d_ws;
  size_t off = 0;
  auto alloc = [&](size_t bytes) -> void* {
    void* p = ws + off;
    off = (off + bytes + 255) & ~(size_t)255;
    return p;
  };
  u16*   whhF   = (u16*)alloc((size_t)G3 * HID * 2);
  u16*   whhB   = (u16*)alloc((size_t)G3 * HID * 2);
  u16*   gwhh16 = (u16*)alloc((size_t)G3 * HID * 2);
  u16*   gwih16 = (u16*)alloc((size_t)G3 * WIH_LD * 2);
  u16*   w1     = (u16*)alloc((size_t)HID * HID * 2);
  u16*   w2     = (u16*)alloc((size_t)HID * HID * 2);
  u16*   aw16   = (u16*)alloc((size_t)N_A * HID * 2);
  float* tgF    = (float*)alloc((size_t)NT * G3 * 4);
  float* tgB    = (float*)alloc((size_t)NT * G3 * 4);
  int*   lines  = (int*)alloc((size_t)NB * N_LINES * 4);
  u16*   hA     = (u16*)alloc((size_t)NB * HID * 2);
  u16*   hB     = (u16*)alloc((size_t)NB * HID * 2);
  u16*   Hcat   = (u16*)alloc((size_t)NB * 1024 * 2);
  float* giH    = (float*)alloc((size_t)NB * G3 * 4);

  const int TPB = 256;
  auto g = [](int n) { return (n + 255) / 256; };

  // one-time weight conversions (f32 -> bf16) and precomputes
  cvt_bf16_kernel<<<g(G3 * HID), TPB, 0, stream>>>(ewhh_f, whhF, G3 * HID);
  cvt_bf16_kernel<<<g(G3 * HID), TPB, 0, stream>>>(ewhh_b, whhB, G3 * HID);
  cvt_bf16_kernel<<<g(G3 * HID), TPB, 0, stream>>>(gwhh, gwhh16, G3 * HID);
  cvt_bf16_kernel<<<g(G3 * WIH_LD), TPB, 0, stream>>>(gwih, gwih16, G3 * WIH_LD);
  cvt_bf16_kernel<<<g(HID * HID), TPB, 0, stream>>>(mlpw, w1, HID * HID);
  cvt_bf16_kernel<<<g(HID * HID), TPB, 0, stream>>>(mlpw + HID * HID, w2, HID * HID);
  cvt_bf16_kernel<<<g(N_A * HID), TPB, 0, stream>>>(actw, aw16, N_A * HID);
  task_gi_kernel<<<g(NT * G3), TPB, 0, stream>>>(embed, ewih_f, ebih_f, tgF);
  task_gi_kernel<<<g(NT * G3), TPB, 0, stream>>>(embed, ewih_b, ebih_b, tgB);
  lines_kernel<<<g(NB * N_LINES), TPB, 0, stream>>>(inputs, lines);

  u16* cur; u16* nxt; u16* tmp;

  // forward encoder (10 steps), gi gathered from task_gi
  fill0_kernel<<<g(NB * HID), TPB, 0, stream>>>(hA, NB * HID);
  cur = hA; nxt = hB;
  for (int s = 0; s < N_LINES; ++s) {
    gru_step_kernel<<<NB / 16, TPB, 0, stream>>>(cur, nxt, whhF, ebhh_f, tgF,
                                                 lines + s, nullptr, nullptr, nullptr);
    tmp = cur; cur = nxt; nxt = tmp;
  }
  copy_half_kernel<<<g(NB * HID), TPB, 0, stream>>>(cur, Hcat, 0);

  // backward encoder (tokens reversed)
  fill0_kernel<<<g(NB * HID), TPB, 0, stream>>>(hA, NB * HID);
  cur = hA; nxt = hB;
  for (int s = 0; s < N_LINES; ++s) {
    gru_step_kernel<<<NB / 16, TPB, 0, stream>>>(cur, nxt, whhB, ebhh_b, tgB,
                                                 lines + (N_LINES - 1 - s), nullptr, nullptr, nullptr);
    tmp = cur; cur = nxt; nxt = tmp;
  }
  copy_half_kernel<<<g(NB * HID), TPB, 0, stream>>>(cur, Hcat, HID);

  // giH = H @ gru_wih[:,64:].T + gru_bih (once; H is constant over t)
  giH_kernel<<<NB / 16, TPB, 0, stream>>>(Hcat, gwih16, gbih, giH);

  // main recurrence: 128 steps x (fused GRU, fused MLP+head)
  h0_kernel<<<g(NB * HID), TPB, 0, stream>>>(rnn_hxs, hA);
  cur = hA; nxt = hB;
  for (int t = 0; t < T_STEPS; ++t) {
    const float* xr = inputs + (size_t)t * NB * IN_ROW;
    gru_step_kernel<<<NB / 16, TPB, 0, stream>>>(cur, nxt, gwhh16, gbhh, giH,
                                                 nullptr, xr, gwih16,
                                                 out + (size_t)t * NB * STATE + 2);
    mlp_head_kernel<<<NB / 16, TPB, 0, stream>>>(nxt, w1, mlpb, w2, mlpb + HID,
                                                 aw16, actb, crw, crb,
                                                 xr + COND + N_LINES,
                                                 out + (size_t)t * NB * STATE, t);
    tmp = cur; cur = nxt; nxt = tmp;
  }

  // hx[-1:] tail
  tail_kernel<<<g(NB * STATE), TPB, 0, stream>>>(out);
}
